// DecoderWithAttention_84310208020587
// MI455X (gfx1250) — matrix operand order repaired and checked by reading, not compiled, not run on hardware
//
#include <hip/hip_runtime.h>
#include <cstdint>
#include <cstddef>

// ---------------- problem constants ----------------
constexpr int NB = 64;      // batch
constexpr int NR = 36;      // regions
constexpr int NF = 2048;    // image feature dim
constexpr int ND = 1024;    // hidden dim
constexpr int NECAP = 1024; // caption emb
constexpr int NEATTR = 512; // attr emb
constexpr int NAV = 512;    // visual att dim
constexpr int NAS = 512;    // semantic att dim
constexpr int NVCAP = 10000;
constexpr int NVCAP_PAD = 10016;  // NVCAP + 16 — weight rows padded for guard-free B tiles
constexpr int NLCAP = 20;
constexpr int NLATTR = 10;
constexpr int NT = 19;      // decode steps = LCAP-1

typedef __attribute__((ext_vector_type(16))) __bf16 v16bf;
typedef __attribute__((ext_vector_type(8)))  float  v8f;

static inline int ceil_div(int a, int b) { return (a + b - 1) / b; }

// ---------------- WMMA bf16 GEMM ----------------
// C[m,n] = sum_k A[m,k]*W[n,k] + bias[n]   (optionally * mask(t < dl[m]))
// A: [M,K] bf16 row-major (M % 16 == 0), W: [Npad,K] bf16 row-major with
// Npad = ceil(N/16)*16 rows present in memory, K % 32 == 0.
// grid = (M/16, ceil(N/128)), block = 256 (8 waves; one 16x16 C tile each)
__global__ void __launch_bounds__(256)
gemm_bf16_wmma(const __bf16* __restrict__ A, const __bf16* __restrict__ W,
               const float* __restrict__ bias, float* __restrict__ C,
               int M, int N, int K, int ldc,
               const int* __restrict__ dl, int tstep)
{
    const int lane = threadIdx.x & 31;
    const int wave = threadIdx.x >> 5;
    const int m0 = blockIdx.x * 16;
    const int n0 = (blockIdx.y * 8 + wave) * 16;
    if (n0 >= N) return;                 // wave-uniform: EXEC stays full for WMMA

    const int hl = lane >> 4;            // lane half: 0 | 1
    const int lm = lane & 15;

    const __bf16* __restrict__ arow = A + (size_t)(m0 + lm) * K;   // M multiple of 16
    const __bf16* __restrict__ wrow = W + (size_t)(n0 + lm) * K;   // rows exist up to Npad

    v8f acc = {};
#pragma unroll 2
    for (int kb = 0; kb < K; kb += 32) {
        const __bf16* ap = arow + kb + hl * 16;   // 32B aligned
        const __bf16* wp = wrow + kb + hl * 16;   // 32B aligned
        __builtin_prefetch((const void*)(wp + 512), 0, 3);
        // B fragment: 32x16 tile; lanes = N cols, halves split K 0..15 / 16..31
        const v16bf b = *(const v16bf*)wp;        // 2x global_load_b128
        // A fragment: contiguous K order matching B: ap[0..15]
        v16bf a;
#pragma unroll
        for (int j = 0; j < 8; ++j) { a[j] = ap[j]; a[j + 8] = ap[j + 8]; }
        acc = __builtin_amdgcn_wmma_f32_16x16x32_bf16(
            false, a, false, b, (short)0, acc, false, false);
    }
    // C store: VGPR i -> row m0 + i + 8*hl, col n0 + lm
#pragma unroll
    for (int i = 0; i < 8; ++i) {
        const int m = m0 + i + hl * 8;
        const int n = n0 + lm;
        if (n < N) {
            float v = acc[i] + (bias ? bias[n] : 0.0f);
            if (dl) v *= (tstep < dl[m]) ? 1.0f : 0.0f;
            C[(size_t)m * ldc + n] = v;
        }
    }
}

// ---------------- prep / elementwise kernels ----------------

// stable descending argsort of lengths (B small); also sorted dec_len
__global__ void sort_lengths_k(const int* __restrict__ cap_len, int* __restrict__ sind,
                               int* __restrict__ dlen, int B)
{
    const int i = threadIdx.x;
    if (i < B) {
        const int li = cap_len[i];
        int rank = 0;
        for (int j = 0; j < B; ++j) {
            const int lj = cap_len[j];
            if (lj > li || (lj == li && j < i)) rank++;
        }
        sind[rank] = i;
    }
    __syncthreads();
    if (i < B) dlen[i] = cap_len[sind[i]] - 1;
}

__global__ void fill_zero_k(float* p, size_t n)
{
    size_t i = (size_t)blockIdx.x * blockDim.x + threadIdx.x;
    if (i < n) p[i] = 0.0f;
}

__global__ void f32_to_bf16_k(const float* __restrict__ s, __bf16* __restrict__ d, size_t n)
{
    size_t i = (size_t)blockIdx.x * blockDim.x + threadIdx.x;
    if (i < n) d[i] = (__bf16)s[i];
}

// pack [Wih | Whh] rows into bf16 [rows][Kx+Kh]
__global__ void pack2_bf16_k(const float* __restrict__ Wih, int Kx,
                             const float* __restrict__ Whh, int Kh,
                             __bf16* __restrict__ dst, int rows)
{
    const int K = Kx + Kh;
    size_t idx = (size_t)blockIdx.x * blockDim.x + threadIdx.x;
    if (idx >= (size_t)rows * K) return;
    const int r = (int)(idx / K);
    const int k = (int)(idx % K);
    const float v = (k < Kx) ? Wih[(size_t)r * Kx + k] : Whh[(size_t)r * Kh + (k - Kx)];
    dst[idx] = (__bf16)v;
}

__global__ void add2_k(const float* a, const float* b, float* c, int n)
{
    int i = blockIdx.x * blockDim.x + threadIdx.x;
    if (i < n) c[i] = a[i] + b[i];
}

// weight-norm fold: dst[r][k] = v[r][k] * g[r] / ||v[r]||
__global__ void wn_pack_k(const float* __restrict__ v, const float* __restrict__ g,
                          __bf16* __restrict__ dst, int rows, int K)
{
    const int r = blockIdx.x * blockDim.x + threadIdx.x;
    if (r >= rows) return;
    const float* row = v + (size_t)r * K;
    float ss = 0.0f;
    for (int k = 0; k < K; ++k) { const float x = row[k]; ss += x * x; }
    const float s = g[r] * rsqrtf(ss);
    __bf16* d = dst + (size_t)r * K;
    for (int k = 0; k < K; ++k) d[k] = (__bf16)(row[k] * s);
}

__global__ void gather_img_k(const float* __restrict__ img, const int* __restrict__ sind,
                             float* __restrict__ img_s, __bf16* __restrict__ img_bf)
{
    size_t idx = (size_t)blockIdx.x * blockDim.x + threadIdx.x;
    const size_t tot = (size_t)NB * NR * NF;
    if (idx >= tot) return;
    const int b = (int)(idx / ((size_t)NR * NF));
    const size_t rem = idx % ((size_t)NR * NF);
    const float v = img[(size_t)sind[b] * NR * NF + rem];
    img_s[idx] = v;
    img_bf[idx] = (__bf16)v;
}

__global__ void img_mean_k(const float* __restrict__ img_s, float* __restrict__ mean)
{
    int idx = blockIdx.x * blockDim.x + threadIdx.x;
    if (idx >= NB * NF) return;
    const int b = idx / NF, f = idx % NF;
    float acc = 0.0f;
    for (int r = 0; r < NR; ++r) acc += img_s[((size_t)b * NR + r) * NF + f];
    mean[idx] = acc * (1.0f / NR);
}

__global__ void attr_mean_k(const float* __restrict__ attr_emb, const int* __restrict__ enc_attr,
                            float* __restrict__ out)
{
    int idx = blockIdx.x * blockDim.x + threadIdx.x;
    if (idx >= NB * NEATTR) return;
    const int b = idx / NEATTR, e = idx % NEATTR;
    float acc = 0.0f;
    for (int l = 0; l < NLATTR; ++l)
        acc += attr_emb[(size_t)enc_attr[b * NLATTR + l] * NEATTR + e];
    out[idx] = acc * (1.0f / NLATTR);
}

__global__ void cap_gather_k(const float* __restrict__ cap_emb, const int* __restrict__ enc_caps,
                             const int* __restrict__ sind, float* __restrict__ out)
{
    size_t idx = (size_t)blockIdx.x * blockDim.x + threadIdx.x;
    const size_t tot = (size_t)NB * NLCAP * NECAP;
    if (idx >= tot) return;
    const int b = (int)(idx / (NLCAP * NECAP));
    const int rem = (int)(idx % (NLCAP * NECAP));
    const int t = rem / NECAP, k = rem % NECAP;
    const int tok = enc_caps[sind[b] * NLCAP + t];
    out[idx] = cap_emb[(size_t)tok * NECAP + k];
}

// 5-way concat (fp32 sources, per-source row stride) -> bf16 [B][K]
__global__ void concat5_bf16_k(
    const float* s0, int n0, int ld0, const float* s1, int n1, int ld1,
    const float* s2, int n2, int ld2, const float* s3, int n3, int ld3,
    const float* s4, int n4, int ld4, __bf16* __restrict__ dst, int B, int K)
{
    size_t idx = (size_t)blockIdx.x * blockDim.x + threadIdx.x;
    if (idx >= (size_t)B * K) return;
    const int b = (int)(idx / K);
    const int k = (int)(idx % K);
    const int c0 = n0, c1 = c0 + n1, c2 = c1 + n2, c3 = c2 + n3;
    float v;
    if      (k < c0) v = s0[(size_t)b * ld0 + k];
    else if (k < c1) v = s1[(size_t)b * ld1 + (k - c0)];
    else if (k < c2) v = s2[(size_t)b * ld2 + (k - c1)];
    else if (k < c3) v = s3[(size_t)b * ld3 + (k - c2)];
    else             v = s4[(size_t)b * ld4 + (k - c3)];
    dst[idx] = (__bf16)v;
}

__device__ __forceinline__ float sigm(float x) { return 1.0f / (1.0f + expf(-x)); }

// gates [B][4D] (i,f,g,o) -> update h,c in place
__global__ void lstm_update_k(const float* __restrict__ g, float* __restrict__ h,
                              float* __restrict__ c, int B, int D)
{
    int idx = blockIdx.x * blockDim.x + threadIdx.x;
    if (idx >= B * D) return;
    const int b = idx / D, d = idx % D;
    const float* gr = g + (size_t)b * 4 * D;
    const float ig = sigm(gr[d]);
    const float fg = sigm(gr[D + d]);
    const float gg = tanhf(gr[2 * D + d]);
    const float og = sigm(gr[3 * D + d]);
    const float c2 = fg * c[idx] + ig * gg;
    c[idx] = c2;
    h[idx] = og * tanhf(c2);
}

// e[b,r] = w . relu(att_img[b,r,:] + hproj[b,:]) + b0
__global__ void vatt_e_k(const float* __restrict__ att_img, const float* __restrict__ hproj,
                         const float* __restrict__ w, const float* __restrict__ b0,
                         float* __restrict__ e)
{
    int idx = blockIdx.x * blockDim.x + threadIdx.x;
    if (idx >= NB * NR) return;
    const int b = idx / NR;
    const float* ai = att_img + (size_t)idx * NAV;
    const float* hp = hproj + (size_t)b * NAV;
    float acc = b0[0];
    for (int a = 0; a < NAV; ++a) {
        const float x = ai[a] + hp[a];
        acc += w[a] * (x > 0.0f ? x : 0.0f);
    }
    e[idx] = acc;
}

// softmax over R + weighted feature sum; one block per batch row
__global__ void vatt_apply_k(const float* __restrict__ e, const float* __restrict__ img_s,
                             float* __restrict__ vatt)
{
    const int b = blockIdx.x;
    __shared__ float sp[64];
    const float* eb = e + b * NR;
    if (threadIdx.x == 0) {
        float m = -1e30f;
        for (int r = 0; r < NR; ++r) m = fmaxf(m, eb[r]);
        float s = 0.0f;
        for (int r = 0; r < NR; ++r) { const float p = expf(eb[r] - m); sp[r] = p; s += p; }
        sp[NR] = 1.0f / s;
    }
    __syncthreads();
    const float inv = sp[NR];
    const float* ib = img_s + (size_t)b * NR * NF;
    for (int f = threadIdx.x; f < NF; f += blockDim.x) {
        float acc = 0.0f;
        for (int r = 0; r < NR; ++r) acc += sp[r] * ib[(size_t)r * NF + f];
        vatt[(size_t)b * NF + f] = acc * inv;
    }
}

// gate[b] = sigmoid(w . relu(a2[b,:]) + b0)
__global__ void satt_gate_k(const float* __restrict__ a2, const float* __restrict__ w,
                            const float* __restrict__ b0, float* __restrict__ gate)
{
    const int b = blockIdx.x * blockDim.x + threadIdx.x;
    if (b >= NB) return;
    const float* ab = a2 + (size_t)b * NAS;
    float acc = b0[0];
    for (int a = 0; a < NAS; ++a) {
        float x = ab[a];
        acc += w[a] * (x > 0.0f ? x : 0.0f);
    }
    gate[b] = sigm(acc);
}

__global__ void satt_apply_k(const float* __restrict__ gate, const float* __restrict__ attr_mean,
                             float* __restrict__ s_att)
{
    int idx = blockIdx.x * blockDim.x + threadIdx.x;
    if (idx >= NB * NEATTR) return;
    s_att[idx] = gate[idx / NEATTR] * attr_mean[idx];
}

// tail: caps (sorted, as float), dec_len, sort_ind
__global__ void tail_k(const int* __restrict__ enc_caps, const int* __restrict__ cap_len,
                       const int* __restrict__ sind, float* __restrict__ o_caps,
                       float* __restrict__ o_dlen, float* __restrict__ o_sind)
{
    int idx = blockIdx.x * blockDim.x + threadIdx.x;
    if (idx < NB * NLCAP) {
        const int b = idx / NLCAP, t = idx % NLCAP;
        o_caps[idx] = (float)enc_caps[sind[b] * NLCAP + t];
    }
    if (idx < NB) {
        o_dlen[idx] = (float)(cap_len[sind[idx]] - 1);
        o_sind[idx] = (float)sind[idx];
    }
}

// ---------------- host launcher ----------------
extern "C" void kernel_launch(void* const* d_in, const int* in_sizes, int n_in,
                              void* d_out, int out_size, void* d_ws, size_t ws_size,
                              hipStream_t stream)
{
    (void)in_sizes; (void)n_in; (void)out_size; (void)ws_size;

    // inputs (setup_inputs order; params dict in insertion order)
    const float* image_features = (const float*)d_in[0];
    const int*   enc_caps       = (const int*)d_in[1];
    const int*   cap_len        = (const int*)d_in[2];
    const int*   enc_attr       = (const int*)d_in[3];
    const float* attr_emb = (const float*)d_in[4];
    const float* cap_emb  = (const float*)d_in[5];
    const float* l1v_Wih = (const float*)d_in[6];
    const float* l1v_Whh = (const float*)d_in[7];
    const float* l1v_bih = (const float*)d_in[8];
    const float* l1v_bhh = (const float*)d_in[9];
    const float* l1s_Wih = (const float*)d_in[10];
    const float* l1s_Whh = (const float*)d_in[11];
    const float* l1s_bih = (const float*)d_in[12];
    const float* l1s_bhh = (const float*)d_in[13];
    const float* l2_Wih  = (const float*)d_in[14];
    const float* l2_Whh  = (const float*)d_in[15];
    const float* l2_bih  = (const float*)d_in[16];
    const float* l2_bhh  = (const float*)d_in[17];
    const float* vatt_Wf = (const float*)d_in[18];
    const float* vatt_bf = (const float*)d_in[19];
    const float* vatt_Wh = (const float*)d_in[20];
    const float* vatt_bh = (const float*)d_in[21];
    const float* vatt_w  = (const float*)d_in[22];
    const float* vatt_b  = (const float*)d_in[23];
    const float* satt_Wx = (const float*)d_in[24];
    const float* satt_bx = (const float*)d_in[25];
    const float* satt_Wh = (const float*)d_in[26];
    const float* satt_bh = (const float*)d_in[27];
    const float* satt_w  = (const float*)d_in[28];
    const float* satt_b  = (const float*)d_in[29];
    const float* fcv_v = (const float*)d_in[30];
    const float* fcv_g = (const float*)d_in[31];
    const float* fcv_b = (const float*)d_in[32];
    const float* fcs_v = (const float*)d_in[33];
    const float* fcs_g = (const float*)d_in[34];
    const float* fcs_b = (const float*)d_in[35];
    const float* fc_v  = (const float*)d_in[36];
    const float* fc_g  = (const float*)d_in[37];
    const float* fc_b  = (const float*)d_in[38];

    // outputs (flat, return order): P, Pv, Ps, caps, dec_len, sort_ind
    float* outP  = (float*)d_out;
    float* outPv = outP  + (size_t)NB * NT * NVCAP;
    float* outPs = outPv + (size_t)NB * NT * NVCAP;
    float* outC  = outPs + (size_t)NB * NT * NVCAP;
    float* outDL = outC + (size_t)NB * NLCAP;
    float* outSI = outDL + NB;

    // workspace bump allocator
    char* base = (char*)d_ws;
    size_t off = 0;
    auto alloc = [&](size_t bytes) -> char* {
        off = (off + 255) & ~(size_t)255;
        char* p = base + off;
        off += bytes;
        return p;
    };

    int*    sind   = (int*)alloc(NB * sizeof(int));
    int*    dlen   = (int*)alloc(NB * sizeof(int));
    // packed bf16 weights (combined Wih|Whh per LSTM)
    const int K1v = ND + NF + NECAP + ND;        // 5120
    const int K1s = ND + NEATTR + NECAP + ND;    // 3584
    const int K2  = NF + ND + NEATTR + ND + ND;  // 5632
    __bf16* Wl1v = (__bf16*)alloc((size_t)4 * ND * K1v * 2);
    __bf16* Wl1s = (__bf16*)alloc((size_t)4 * ND * K1s * 2);
    __bf16* Wl2  = (__bf16*)alloc((size_t)4 * ND * K2  * 2);
    float*  bl1v = (float*)alloc(4 * ND * 4);
    float*  bl1s = (float*)alloc(4 * ND * 4);
    float*  bl2  = (float*)alloc(4 * ND * 4);
    // vocab projections padded to NVCAP_PAD rows (guard-free 16x16 B tiles)
    __bf16* Wfcv = (__bf16*)alloc((size_t)NVCAP_PAD * ND * 2);
    __bf16* Wfcs = (__bf16*)alloc((size_t)NVCAP_PAD * ND * 2);
    __bf16* Wfc  = (__bf16*)alloc((size_t)NVCAP_PAD * ND * 2);
    __bf16* Wvf  = (__bf16*)alloc((size_t)NAV * NF * 2);
    __bf16* Wvh  = (__bf16*)alloc((size_t)NAV * ND * 2);
    const int Ks = NEATTR + ND;                  // 1536
    __bf16* Wsat = (__bf16*)alloc((size_t)NAS * Ks * 2);
    float*  bsat = (float*)alloc(NAS * 4);
    // activations
    float*  img_s    = (float*)alloc((size_t)NB * NR * NF * 4);
    __bf16* img_bf   = (__bf16*)alloc((size_t)NB * NR * NF * 2);
    float*  img_mean = (float*)alloc((size_t)NB * NF * 4);
    float*  attr_m   = (float*)alloc((size_t)NB * NEATTR * 4);
    float*  capemb_s = (float*)alloc((size_t)NB * NLCAP * NECAP * 4);
    float*  att_img  = (float*)alloc((size_t)NB * NR * NAV * 4);
    float*  states   = (float*)alloc((size_t)6 * NB * ND * 4);
    float *h1v = states, *c1v = h1v + NB * ND, *h1s = c1v + NB * ND,
          *c1s = h1s + NB * ND, *h2 = c1s + NB * ND, *c2 = h2 + NB * ND;
    float*  gates  = (float*)alloc((size_t)NB * 4 * ND * 4);
    __bf16* xcat   = (__bf16*)alloc((size_t)NB * K2 * 2);
    __bf16* hbf    = (__bf16*)alloc((size_t)NB * ND * 2);
    float*  hproj  = (float*)alloc((size_t)NB * NAV * 4);
    float*  ebuf   = (float*)alloc((size_t)NB * NR * 4);
    float*  v_att  = (float*)alloc((size_t)NB * NF * 4);
    float*  a2buf  = (float*)alloc((size_t)NB * NAS * 4);
    float*  sgate  = (float*)alloc(NB * 4);
    float*  s_att  = (float*)alloc((size_t)NB * NEATTR * 4);

    auto gemm = [&](const __bf16* A, const __bf16* W, const float* bias, float* C,
                    int M, int N, int K, int ldc, const int* dl, int t) {
        dim3 g(ceil_div(M, 16), ceil_div(N, 128));
        gemm_bf16_wmma<<<g, 256, 0, stream>>>(A, W, bias, C, M, N, K, ldc, dl, t);
    };
    auto launch1d = [&](size_t n) { return dim3((unsigned)((n + 255) / 256)); };

    // ---- prep (once per call; deterministic) ----
    sort_lengths_k<<<1, NB, 0, stream>>>(cap_len, sind, dlen, NB);
    pack2_bf16_k<<<launch1d((size_t)4 * ND * K1v), 256, 0, stream>>>(l1v_Wih, K1v - ND, l1v_Whh, ND, Wl1v, 4 * ND);
    pack2_bf16_k<<<launch1d((size_t)4 * ND * K1s), 256, 0, stream>>>(l1s_Wih, K1s - ND, l1s_Whh, ND, Wl1s, 4 * ND);
    pack2_bf16_k<<<launch1d((size_t)4 * ND * K2),  256, 0, stream>>>(l2_Wih,  K2  - ND, l2_Whh,  ND, Wl2,  4 * ND);
    add2_k<<<launch1d(4 * ND), 256, 0, stream>>>(l1v_bih, l1v_bhh, bl1v, 4 * ND);
    add2_k<<<launch1d(4 * ND), 256, 0, stream>>>(l1s_bih, l1s_bhh, bl1s, 4 * ND);
    add2_k<<<launch1d(4 * ND), 256, 0, stream>>>(l2_bih,  l2_bhh,  bl2,  4 * ND);
    wn_pack_k<<<ceil_div(NVCAP, 256), 256, 0, stream>>>(fcv_v, fcv_g, Wfcv, NVCAP, ND);
    wn_pack_k<<<ceil_div(NVCAP, 256), 256, 0, stream>>>(fcs_v, fcs_g, Wfcs, NVCAP, ND);
    wn_pack_k<<<ceil_div(NVCAP, 256), 256, 0, stream>>>(fc_v,  fc_g,  Wfc,  NVCAP, ND);
    // zero the pad rows of the vocab weights (bf16 pad region as floats)
    {
        const size_t padf = (size_t)(NVCAP_PAD - NVCAP) * ND / 2; // bf16 pairs as f32
        fill_zero_k<<<launch1d(padf), 256, 0, stream>>>((float*)(Wfcv + (size_t)NVCAP * ND), padf);
        fill_zero_k<<<launch1d(padf), 256, 0, stream>>>((float*)(Wfcs + (size_t)NVCAP * ND), padf);
        fill_zero_k<<<launch1d(padf), 256, 0, stream>>>((float*)(Wfc  + (size_t)NVCAP * ND), padf);
    }
    f32_to_bf16_k<<<launch1d((size_t)NAV * NF), 256, 0, stream>>>(vatt_Wf, Wvf, (size_t)NAV * NF);
    f32_to_bf16_k<<<launch1d((size_t)NAV * ND), 256, 0, stream>>>(vatt_Wh, Wvh, (size_t)NAV * ND);
    pack2_bf16_k<<<launch1d((size_t)NAS * Ks), 256, 0, stream>>>(satt_Wx, NEATTR, satt_Wh, ND, Wsat, NAS);
    add2_k<<<launch1d(NAS), 256, 0, stream>>>(satt_bx, satt_bh, bsat, NAS);
    gather_img_k<<<launch1d((size_t)NB * NR * NF), 256, 0, stream>>>(image_features, sind, img_s, img_bf);
    img_mean_k<<<launch1d((size_t)NB * NF), 256, 0, stream>>>(img_s, img_mean);
    attr_mean_k<<<launch1d((size_t)NB * NEATTR), 256, 0, stream>>>(attr_emb, enc_attr, attr_m);
    cap_gather_k<<<launch1d((size_t)NB * NLCAP * NECAP), 256, 0, stream>>>(cap_emb, enc_caps, sind, capemb_s);
    fill_zero_k<<<launch1d((size_t)6 * NB * ND), 256, 0, stream>>>(states, (size_t)6 * NB * ND);
    // hoisted: att_img[b,r,:] = img @ vatt_Wf^T + vatt_bf  (time-invariant)
    gemm(img_bf, Wvf, vatt_bf, att_img, NB * NR, NAV, NF, NAV, nullptr, 0);

    const int ldP = NT * NVCAP;
    for (int t = 0; t < NT; ++t) {
        const float* embt = capemb_s + (size_t)t * NECAP;
        // --- LSTM 1v: A = [h2, img_mean, emb_t, h1v]
        concat5_bf16_k<<<launch1d((size_t)NB * K1v), 256, 0, stream>>>(
            h2, ND, ND, img_mean, NF, NF, embt, NECAP, NLCAP * NECAP,
            h1v, ND, ND, h2, 0, 0, xcat, NB, K1v);
        gemm(xcat, Wl1v, bl1v, gates, NB, 4 * ND, K1v, 4 * ND, nullptr, 0);
        lstm_update_k<<<launch1d(NB * ND), 256, 0, stream>>>(gates, h1v, c1v, NB, ND);
        f32_to_bf16_k<<<launch1d((size_t)NB * ND), 256, 0, stream>>>(h1v, hbf, (size_t)NB * ND);
        // --- visual attention
        gemm(hbf, Wvh, vatt_bh, hproj, NB, NAV, ND, NAV, nullptr, 0);
        vatt_e_k<<<launch1d(NB * NR), 256, 0, stream>>>(att_img, hproj, vatt_w, vatt_b, ebuf);
        vatt_apply_k<<<NB, 256, 0, stream>>>(ebuf, img_s, v_att);
        // --- preds_v (masked)
        gemm(hbf, Wfcv, fcv_b, outPv + (size_t)t * NVCAP, NB, NVCAP, ND, ldP, dlen, t);
        // --- LSTM 1s: A = [h2, attr_mean, emb_t, h1s]
        concat5_bf16_k<<<launch1d((size_t)NB * K1s), 256, 0, stream>>>(
            h2, ND, ND, attr_m, NEATTR, NEATTR, embt, NECAP, NLCAP * NECAP,
            h1s, ND, ND, h2, 0, 0, xcat, NB, K1s);
        gemm(xcat, Wl1s, bl1s, gates, NB, 4 * ND, K1s, 4 * ND, nullptr, 0);
        lstm_update_k<<<launch1d(NB * ND), 256, 0, stream>>>(gates, h1s, c1s, NB, ND);
        // --- semantic attention gate
        concat5_bf16_k<<<launch1d((size_t)NB * Ks), 256, 0, stream>>>(
            attr_m, NEATTR, NEATTR, h1s, ND, ND, h2, 0, 0, h2, 0, 0, h2, 0, 0, xcat, NB, Ks);
        gemm(xcat, Wsat, bsat, a2buf, NB, NAS, Ks, NAS, nullptr, 0);
        satt_gate_k<<<1, NB, 0, stream>>>(a2buf, satt_w, satt_b, sgate);
        satt_apply_k<<<launch1d(NB * NEATTR), 256, 0, stream>>>(sgate, attr_m, s_att);
        // --- preds_s (masked)
        f32_to_bf16_k<<<launch1d((size_t)NB * ND), 256, 0, stream>>>(h1s, hbf, (size_t)NB * ND);
        gemm(hbf, Wfcs, fcs_b, outPs + (size_t)t * NVCAP, NB, NVCAP, ND, ldP, dlen, t);
        // --- LSTM 2: A = [v_att, h1v, s_att, h1s, h2]
        concat5_bf16_k<<<launch1d((size_t)NB * K2), 256, 0, stream>>>(
            v_att, NF, NF, h1v, ND, ND, s_att, NEATTR, NEATTR,
            h1s, ND, ND, h2, ND, ND, xcat, NB, K2);
        gemm(xcat, Wl2, bl2, gates, NB, 4 * ND, K2, 4 * ND, nullptr, 0);
        lstm_update_k<<<launch1d(NB * ND), 256, 0, stream>>>(gates, h2, c2, NB, ND);
        // --- preds (masked)
        f32_to_bf16_k<<<launch1d((size_t)NB * ND), 256, 0, stream>>>(h2, hbf, (size_t)NB * ND);
        gemm(hbf, Wfc, fc_b, outP + (size_t)t * NVCAP, NB, NVCAP, ND, ldP, dlen, t);
    }

    tail_k<<<ceil_div(NB * NLCAP, 256), 256, 0, stream>>>(enc_caps, cap_len, sind, outC, outDL, outSI);
}
